// TemporalAttention_78623671320764
// MI455X (gfx1250) — matrix operand order repaired and checked by reading, not compiled, not run on hardware
//
#include <hip/hip_runtime.h>

// Problem constants
#define BB   2
#define CC   512
#define TT   16
#define HWP  576            // H*W = 24*24
#define NPOS 18432          // B*T*HW
#define NHEAD 8
#define HD   64

typedef __attribute__((ext_vector_type(16))) _Float16 v16h;
typedef __attribute__((ext_vector_type(8)))  float    v8f;
typedef __attribute__((ext_vector_type(4)))  unsigned int v4u;
typedef __attribute__((ext_vector_type(8)))  int      v8i;
typedef __attribute__((ext_vector_type(4)))  int      v4i;

// ---------------------------------------------------------------------------
// WMMA fragment helpers (CDNA5 gfx1250, wave32)
// A 16x32 f16: lane(m=l&15, h=l>>4): halfs[0..7]=A[m][k+h*8..], halfs[8..15]=A[m][k+16+h*8..]
// B 32x16 f16: lane(n=l&15, g=l>>4): halfs[0..7]=B[k+g*8..][n], halfs[8..15]=B[k+16+g*8..][n]
// C/D 16x16 f32: lane(n=l&15, hi=l>>4): vgpr j holds D[hi*8+j][n]
// ---------------------------------------------------------------------------
__device__ __forceinline__ v16h make_frag(const _Float16* p0, const _Float16* p1) {
  v16h f;
  ((uint4*)&f)[0] = *(const uint4*)p0;
  ((uint4*)&f)[1] = *(const uint4*)p1;
  return f;
}
// base points at Elem(row0, kBase) of a row-major [.. x stride] matrix
__device__ __forceinline__ v16h fragA(const _Float16* base, int stride, int lane) {
  const _Float16* p = base + (lane & 15) * stride + (lane >> 4) * 8;
  return make_frag(p, p + 16);
}
// base points at Elem(col0-as-row, kBase): B[k][n] sourced from row-major M[n][k]
__device__ __forceinline__ v16h fragB(const _Float16* base, int stride, int lane) {
  const _Float16* p = base + (lane & 15) * stride + (lane >> 4) * 8;
  return make_frag(p, p + 16);
}
__device__ __forceinline__ v8f wmma_f16(v16h a, v16h b, v8f c) {
  return __builtin_amdgcn_wmma_f32_16x16x32_f16(false, a, false, b, (short)0, c,
                                                false, false);
}

// ---------------------------------------------------------------------------
// prep_x: xh[r][c] = f16( x[b][c][t][hw] + pos[c][t] ),  r = (b*T+t)*HW + hw
// ---------------------------------------------------------------------------
__global__ void prep_x(const float* __restrict__ x, const float* __restrict__ pos,
                       _Float16* __restrict__ xh) {
  int idx = blockIdx.x * blockDim.x + threadIdx.x;   // over B*C*T*HW, x-layout order
  if (idx >= BB * CC * TT * HWP) return;
  int hw = idx % HWP;
  int tmp = idx / HWP;
  int t = tmp & (TT - 1);  tmp >>= 4;
  int c = tmp & (CC - 1);
  int b = tmp >> 9;
  float v = x[idx] + pos[c * TT + t];
  int r = (b * TT + t) * HWP + hw;
  xh[(size_t)r * CC + c] = (_Float16)v;
}

// ---------------------------------------------------------------------------
// prep_w: wqkv[1536][512] = f16 concat(wq,wk,wv); woh = f16 wo; bqkv = concat bias
// ---------------------------------------------------------------------------
__global__ void prep_w(const float* __restrict__ wq, const float* __restrict__ wk,
                       const float* __restrict__ wv, const float* __restrict__ wo,
                       const float* __restrict__ bq, const float* __restrict__ bk,
                       const float* __restrict__ bv,
                       _Float16* __restrict__ wqkv, _Float16* __restrict__ woh,
                       float* __restrict__ bqkv) {
  int idx = blockIdx.x * blockDim.x + threadIdx.x;
  const int NW = 3 * CC * CC;            // 786432
  const int NWO = CC * CC;               // 262144
  if (idx < NW) {
    int o = idx >> 9, k = idx & (CC - 1);
    const float* src = (o < CC) ? wq : (o < 2 * CC ? wk : wv);
    wqkv[idx] = (_Float16)src[(o & (CC - 1)) * CC + k];
  } else if (idx < NW + NWO) {
    int j = idx - NW;
    woh[j] = (_Float16)wo[j];
  } else if (idx < NW + NWO + 3 * CC) {
    int j = idx - NW - NWO;
    bqkv[j] = (j < CC) ? bq[j] : (j < 2 * CC ? bk[j - CC] : bv[j - 2 * CC]);
  }
}

// ---------------------------------------------------------------------------
// gemm_qkv: Y[18432][1536] f16 = X[18432][512] * W[1536][512]^T + bias
// block = 128 thr (4 waves); wave tile 32x64; wg tile 128x64
// ---------------------------------------------------------------------------
__global__ __launch_bounds__(128)
void gemm_qkv(const _Float16* __restrict__ X, const _Float16* __restrict__ W,
              const float* __restrict__ bias, _Float16* __restrict__ Y) {
  const int lane = threadIdx.x & 31, wave = threadIdx.x >> 5;
  const int rBase = blockIdx.x * 128 + wave * 32;
  const int oBase = blockIdx.y * 64;
  v8f acc[2][4] = {};
#pragma unroll 2
  for (int k = 0; k < CC; k += 32) {
    v16h a[2], b[4];
#pragma unroll
    for (int mi = 0; mi < 2; ++mi)
      a[mi] = fragA(X + (size_t)(rBase + mi * 16) * CC + k, CC, lane);
#pragma unroll
    for (int ni = 0; ni < 4; ++ni)
      b[ni] = fragB(W + (size_t)(oBase + ni * 16) * CC + k, CC, lane);
#pragma unroll
    for (int mi = 0; mi < 2; ++mi)
#pragma unroll
      for (int ni = 0; ni < 4; ++ni)
        acc[mi][ni] = wmma_f16(a[mi], b[ni], acc[mi][ni]);
  }
  const int n_l = lane & 15, g = lane >> 4;
#pragma unroll
  for (int ni = 0; ni < 4; ++ni) {
    int o = oBase + ni * 16 + n_l;
    float bz = bias[o];
#pragma unroll
    for (int mi = 0; mi < 2; ++mi)
#pragma unroll
      for (int j = 0; j < 8; ++j) {
        int r = rBase + mi * 16 + g * 8 + j;
        Y[(size_t)r * (3 * CC) + o] = (_Float16)(acc[mi][ni][j] + bz);
      }
  }
}

// ---------------------------------------------------------------------------
// attn: one workgroup per (b*T+t, head).  QKV row-major [18432][1536],
// head h cols: Q=[h*64..), K=512+h*64, V=1024+h*64.
// LDS: K[576][64] (staged by Tensor Data Mover) + V^T[64][576] + per-wave P slab.
// Flash-style streaming softmax over 64-key blocks; all matmuls via WMMA.
// O written f16 row-major [18432][512].
// ---------------------------------------------------------------------------
__global__ __launch_bounds__(288)
void attn_kernel(const _Float16* __restrict__ QKV, _Float16* __restrict__ O) {
  extern __shared__ _Float16 smem[];
  _Float16* Ksm = smem;                       // [576][64]
  _Float16* Vt  = smem + HWP * HD;            // [64][576]
  const int inst = blockIdx.x;                // 0..255
  const int bt = inst >> 3, head = inst & 7;
  const int tid = threadIdx.x, lane = tid & 31, wave = tid >> 5;
  _Float16* pSm = smem + 2 * HWP * HD + wave * (16 * HD);
  const size_t rowStart = (size_t)bt * HWP;
  const int LDQ = 3 * CC;

  // ---- stage K via Tensor Data Mover: one 2D tile DMA (global->LDS) ----
  // D# group0: count=1 | lds_addr | global_addr[56:0] | type=2
  // D# group1: data_size=2B, tensor_dim0=64, tensor_dim1=576,
  //            tile_dim0=64, tile_dim1=576, tensor_dim0_stride=1536
  if (tid < 32) {
    const _Float16* kGlob = QKV + rowStart * LDQ + CC + head * HD;
    unsigned long long ga = (unsigned long long)(uintptr_t)kGlob;
    v4u g0;
    g0[0] = 1u;                                        // count=1, user mode
    g0[1] = (unsigned)(uintptr_t)Ksm;                  // lds_addr (bytes)
    g0[2] = (unsigned)(ga & 0xffffffffu);              // global_addr[31:0]
    g0[3] = (unsigned)((ga >> 32) & 0x01ffffffu) | 0x80000000u;  // [56:32]|type=2
    v8i g1;
    g1[0] = (int)(1u << 16);       // wg_mask=0 (no cluster), data_size=1 (2 bytes)
    g1[1] = (int)(64u << 16);      // tensor_dim0 = 64  (bits 63:48)
    g1[2] = (int)(576u << 16);     // tensor_dim1 = 576 (bits 95:80)
    g1[3] = (int)(64u << 16);      // tile_dim0 = 64    (bits 127:112)
    g1[4] = (int)576u;             // tile_dim1 = 576   (bits 143:128)
    g1[5] = (int)(3 * CC);         // tensor_dim0_stride = 1536 (bits 191:160)
    g1[6] = 0;
    g1[7] = 0;
    v4i gz4 = {0, 0, 0, 0};
    v8i gz8 = {0, 0, 0, 0, 0, 0, 0, 0};
    __builtin_amdgcn_tensor_load_to_lds(g0, g1, gz4, gz4, gz8, 0);
  }

  // ---- stage V transposed into LDS (TDM cannot transpose) ----
  for (int i = tid; i < HWP * HD / 8; i += 288) {
    int row = i >> 3;             // key index 0..575
    int col = (i & 7) * 8;        // d offset 0..56
    uint4 vq = *(const uint4*)(QKV + (rowStart + row) * LDQ + 2 * CC +
                               head * HD + col);
    const _Float16* vh = (const _Float16*)&vq;
#pragma unroll
    for (int d = 0; d < 8; ++d) Vt[(col + d) * HWP + row] = vh[d];
  }
  if (tid < 32) __builtin_amdgcn_s_wait_tensorcnt(0);   // K tile landed
  __syncthreads();

  // ---- per wave: 4 Q row-tiles of 16 (wave w owns rows [w*64, w*64+64)) ----
#pragma unroll 1
  for (int qt = 0; qt < 4; ++qt) {
    const int qRow = wave * 64 + qt * 16;
    const _Float16* Qbase = QKV + (rowStart + qRow) * LDQ + head * HD;
    v16h aq0 = fragA(Qbase, LDQ, lane);
    v16h aq1 = fragA(Qbase + 32, LDQ, lane);

    float mst[8], lst[8];
    v8f oacc[4] = {};
#pragma unroll
    for (int j = 0; j < 8; ++j) { mst[j] = -1e30f; lst[j] = 0.f; }

#pragma unroll 1
    for (int jb = 0; jb < 9; ++jb) {           // 9 blocks of 64 keys
      const int keyBase = jb * 64;
      v8f s[4] = {};
#pragma unroll
      for (int ni = 0; ni < 4; ++ni) {
        const _Float16* kb = Ksm + (keyBase + ni * 16) * HD;
        s[ni] = wmma_f16(aq0, fragB(kb, HD, lane), s[ni]);
        s[ni] = wmma_f16(aq1, fragB(kb + 32, HD, lane), s[ni]);
      }
      // streaming softmax update (rows live per lane-group, 8 rows each)
      const float scale = 0.125f;              // 64^-0.5
#pragma unroll
      for (int j = 0; j < 8; ++j) {
        float v = fmaxf(fmaxf(s[0][j], s[1][j]), fmaxf(s[2][j], s[3][j])) * scale;
#pragma unroll
        for (int off = 8; off >= 1; off >>= 1) v = fmaxf(v, __shfl_xor(v, off, 32));
        float nm = fmaxf(mst[j], v);
        float alpha = __expf(mst[j] - nm);
        mst[j] = nm;
        float rsum = 0.f;
#pragma unroll
        for (int ni = 0; ni < 4; ++ni) {
          float p = __expf(s[ni][j] * scale - nm);
          s[ni][j] = p;
          rsum += p;
        }
#pragma unroll
        for (int off = 8; off >= 1; off >>= 1) rsum += __shfl_xor(rsum, off, 32);
        lst[j] = lst[j] * alpha + rsum;
#pragma unroll
        for (int dt = 0; dt < 4; ++dt) oacc[dt][j] *= alpha;
      }
      // D-layout -> A-layout via per-wave LDS bounce
      {
        const int n_l = lane & 15, g = lane >> 4;
#pragma unroll
        for (int ni = 0; ni < 4; ++ni)
#pragma unroll
          for (int j = 0; j < 8; ++j)
            pSm[(g * 8 + j) * HD + ni * 16 + n_l] = (_Float16)s[ni][j];
      }
      asm volatile("s_wait_dscnt 0x0" ::: "memory");
      v16h ap0 = fragA(pSm, HD, lane);
      v16h ap1 = fragA(pSm + 32, HD, lane);
#pragma unroll
      for (int dt = 0; dt < 4; ++dt) {
        const _Float16* vb = Vt + (dt * 16) * HWP + keyBase;
        oacc[dt] = wmma_f16(ap0, fragB(vb, HWP, lane), oacc[dt]);
        oacc[dt] = wmma_f16(ap1, fragB(vb + 32, HWP, lane), oacc[dt]);
      }
    }
    // normalize + store O f16 row-major [r][512]
    const int n_l = lane & 15, g = lane >> 4;
#pragma unroll
    for (int dt = 0; dt < 4; ++dt)
#pragma unroll
      for (int j = 0; j < 8; ++j) {
        size_t r = rowStart + qRow + g * 8 + j;
        O[r * CC + head * HD + dt * 16 + n_l] =
            (_Float16)(oacc[dt][j] * (1.f / lst[j]));
      }
  }
}

// ---------------------------------------------------------------------------
// gemm_proj: out[b][o][t][hw] f32 = Oin[18432][512] * Wo[512][512]^T + bo
// ---------------------------------------------------------------------------
__global__ __launch_bounds__(128)
void gemm_proj(const _Float16* __restrict__ A, const _Float16* __restrict__ W,
               const float* __restrict__ bias, float* __restrict__ out) {
  const int lane = threadIdx.x & 31, wave = threadIdx.x >> 5;
  const int rBase = blockIdx.x * 128 + wave * 32;
  const int oBase = blockIdx.y * 64;
  v8f acc[2][4] = {};
#pragma unroll 2
  for (int k = 0; k < CC; k += 32) {
    v16h a[2], b[4];
#pragma unroll
    for (int mi = 0; mi < 2; ++mi)
      a[mi] = fragA(A + (size_t)(rBase + mi * 16) * CC + k, CC, lane);
#pragma unroll
    for (int ni = 0; ni < 4; ++ni)
      b[ni] = fragB(W + (size_t)(oBase + ni * 16) * CC + k, CC, lane);
#pragma unroll
    for (int mi = 0; mi < 2; ++mi)
#pragma unroll
      for (int ni = 0; ni < 4; ++ni)
        acc[mi][ni] = wmma_f16(a[mi], b[ni], acc[mi][ni]);
  }
  const int n_l = lane & 15, g = lane >> 4;
#pragma unroll
  for (int ni = 0; ni < 4; ++ni) {
    int o = oBase + ni * 16 + n_l;
    float bz = bias[o];
#pragma unroll
    for (int mi = 0; mi < 2; ++mi)
#pragma unroll
      for (int j = 0; j < 8; ++j) {
        int r = rBase + mi * 16 + g * 8 + j;
        int b = r / (TT * HWP);
        int rem = r - b * (TT * HWP);
        int t = rem / HWP;
        int hw = rem - t * HWP;
        out[(((size_t)b * CC + o) * TT + t) * HWP + hw] = acc[mi][ni][j] + bz;
      }
  }
}

// ---------------------------------------------------------------------------
extern "C" void kernel_launch(void* const* d_in, const int* in_sizes, int n_in,
                              void* d_out, int out_size, void* d_ws, size_t ws_size,
                              hipStream_t stream) {
  const float* x   = (const float*)d_in[0];
  const float* wq  = (const float*)d_in[1];
  const float* bq  = (const float*)d_in[2];
  const float* wk  = (const float*)d_in[3];
  const float* bk  = (const float*)d_in[4];
  const float* wv  = (const float*)d_in[5];
  const float* bv  = (const float*)d_in[6];
  const float* wo  = (const float*)d_in[7];
  const float* bo  = (const float*)d_in[8];
  const float* pos = (const float*)d_in[9];

  char* ws = (char*)d_ws;
  size_t off = 0;
  _Float16* xh   = (_Float16*)(ws + off); off += (size_t)NPOS * CC * 2;        // 18.9 MB
  _Float16* wqkv = (_Float16*)(ws + off); off += (size_t)3 * CC * CC * 2;      // 1.6 MB
  _Float16* woh  = (_Float16*)(ws + off); off += (size_t)CC * CC * 2;          // 0.5 MB
  float*    bqkv = (float*)(ws + off);    off += (size_t)3 * CC * 4 + 2048;    // bias
  _Float16* qkv  = (_Float16*)(ws + off); off += (size_t)NPOS * 3 * CC * 2;    // 56.6 MB
  _Float16* obuf = (_Float16*)(ws + off); off += (size_t)NPOS * CC * 2;        // 18.9 MB

  // 1. x + pos_emb -> f16, transposed to [pos][channel]
  {
    int n = BB * CC * TT * HWP;
    prep_x<<<(n + 255) / 256, 256, 0, stream>>>(x, pos, xh);
  }
  // 2. weight/bias conversion
  {
    int n = 3 * CC * CC + CC * CC + 3 * CC;
    prep_w<<<(n + 255) / 256, 256, 0, stream>>>(wq, wk, wv, wo, bq, bk, bv,
                                                wqkv, woh, bqkv);
  }
  // 3. fused QKV projection (WMMA f16)
  gemm_qkv<<<dim3(NPOS / 128, (3 * CC) / 64), 128, 0, stream>>>(xh, wqkv, bqkv, qkv);

  // 4. attention, one WG per (b,t,head); 165,888 B dynamic LDS (needs >64KB cap)
  const int attn_lds = (2 * HWP * HD + 9 * 16 * HD) * 2;
  (void)hipFuncSetAttribute((const void*)attn_kernel,
                            hipFuncAttributeMaxDynamicSharedMemorySize, attn_lds);
  attn_kernel<<<BB * TT * NHEAD, 288, attn_lds, stream>>>(qkv, obuf);

  // 5. output projection -> f32 [B,C,T,H,W] + bias
  gemm_proj<<<dim3(NPOS / 128, CC / 64), 128, 0, stream>>>(obuf, woh, bo,
                                                           (float*)d_out);
}
